// GCNNetwork_1271310320020
// MI455X (gfx1250) — compile-verified, matrix operand order checked by reading, not run on hardware
//
#include <hip/hip_runtime.h>
#include <hip/hip_bf16.h>

typedef __attribute__((ext_vector_type(16))) __bf16 v16bf;
typedef __attribute__((ext_vector_type(8)))  __bf16 v8bf;
typedef __attribute__((ext_vector_type(8)))  float  v8f;

#define DD 128
#define SPITCH 136   // bf16 elements per LDS row (128 + 8 pad); 272B row stride, 16B aligned
#define CEILDIV(a,b) (((a)+(b)-1)/(b))

// ---------------- utility kernels ----------------

__global__ void k_fill(float* __restrict__ p, float v, int n) {
  int i = blockIdx.x * blockDim.x + threadIdx.x;
  if (i < n) p[i] = v;
}

// Convert one 128x128 f32 weight matrix into bf16 in WMMA B-fragment order:
// flat = ((colTile*4 + kStep)*32 + lane)*16 + j
// lane: half = lane>>4, l16 = lane&15, kb = half*8
// j<8 : element (k = kStep*32 + kb + j,      col = colTile*16 + l16)
// j>=8: element (k = kStep*32 + kb + 8 + j,  col = colTile*16 + l16)   // kb+16+(j-8)
__global__ void k_cvt_swz(const float* __restrict__ in, __bf16* __restrict__ out) {
  int i = blockIdx.x * blockDim.x + threadIdx.x;   // [0, 16384)
  if (i >= DD * DD) return;
  int j   = i & 15;
  int l   = (i >> 4) & 31;
  int s   = (i >> 9) & 3;
  int t   = i >> 11;
  int half = l >> 4, l16 = l & 15;
  int k = s * 32 + half * 8 + ((j < 8) ? j : (j + 8));
  int c = t * 16 + l16;
  out[i] = (__bf16)in[k * DD + c];
}

// ---------------- degree / norm ----------------

__global__ void k_scatter_deg(const int* __restrict__ dst, const float* __restrict__ w,
                              float* __restrict__ deg, int E) {
  int i = blockIdx.x * blockDim.x + threadIdx.x;
  if (i < E) atomicAdd(&deg[dst[i]], w[i]);
}

__global__ void k_rsqrt(float* __restrict__ p, int n) {
  int i = blockIdx.x * blockDim.x + threadIdx.x;
  if (i < n) p[i] = rsqrtf(fmaxf(p[i], 1e-5f));   // deg >= 2 (self-loop fill 2.0)
}

__global__ void k_norm(const int* __restrict__ src, const int* __restrict__ dst,
                       const float* __restrict__ w, const float* __restrict__ dinv,
                       float* __restrict__ norm, float* __restrict__ selfnorm, int E, int N) {
  int i = blockIdx.x * blockDim.x + threadIdx.x;
  if (i < E) {
    norm[i] = dinv[src[i]] * w[i] * dinv[dst[i]];
  } else if (i < E + N) {
    int v = i - E;
    float d = dinv[v];
    selfnorm[v] = 2.0f * d * d;                   // self-loop weight 2.0
  }
}

// ---------------- WMMA GEMM: C[M,128] = A[M,128] @ W[128,128] (+bias) ----------------
// block = 256 threads (8 waves); block computes 16 rows x 128 cols, wave w owns cols [16w,16w+16).
// Wsw is bf16 in B-fragment order (see k_cvt_swz): each lane reads its v16bf as one
// contiguous 32B chunk from global (L2-resident, reused across all blocks).

__global__ __launch_bounds__(256) void k_gemm_wmma(
    const float* __restrict__ A, const __bf16* __restrict__ Wsw,
    float* __restrict__ C, const float* __restrict__ bias, int M)
{
  __shared__ __bf16 sA[16 * SPITCH];

  int tid = threadIdx.x;
  int mBase = blockIdx.x * 16;

  // stage A tile (16x128 f32 -> bf16): one thread = 8 contiguous elements
  {
    int base = tid * 8;                 // 256 threads * 8 = 2048 = 16*128
    int r = base >> 7, c = base & 127;
    int gm = mBase + r;
    v8bf tbuf = {};
    if (gm < M) {
      const float4* q = (const float4*)(A + (size_t)gm * DD + c);
      float4 u0 = q[0], u1 = q[1];
      tbuf[0] = (__bf16)u0.x; tbuf[1] = (__bf16)u0.y;
      tbuf[2] = (__bf16)u0.z; tbuf[3] = (__bf16)u0.w;
      tbuf[4] = (__bf16)u1.x; tbuf[5] = (__bf16)u1.y;
      tbuf[6] = (__bf16)u1.z; tbuf[7] = (__bf16)u1.w;
    }
    *(v8bf*)(sA + r * SPITCH + c) = tbuf;
  }
  __syncthreads();

  int wave = tid >> 5, lane = tid & 31;
  int half = lane >> 4, l16 = lane & 15;
  int kb = half * 8;

  const v16bf* Wf = (const v16bf*)Wsw + ((size_t)wave * 4 * 32 + lane);

  v8f acc = {};
  #pragma unroll
  for (int s = 0; s < 4; ++s) {
    const v8bf* pa = (const v8bf*)(sA + l16 * SPITCH + s * 32 + kb);
    v8bf lo = pa[0];        // K = s*32 + kb + 0..7
    v8bf hi = pa[2];        // K = s*32 + kb + 16..23
    v16bf a = __builtin_shufflevector(lo, hi, 0, 1, 2, 3, 4, 5, 6, 7,
                                              8, 9, 10, 11, 12, 13, 14, 15);
    v16bf b = Wf[s * 32];   // contiguous 32B fragment
    acc = __builtin_amdgcn_wmma_f32_16x16x32_bf16(false, a, false, b,
                                                  (short)0, acc, false, false);
  }

  int nBase = wave * 16;
  float bv = bias ? bias[nBase + l16] : 0.0f;
  // D layout: VGPR r -> row (mBase + r + half*8), col (nBase + l16)
  if (mBase + 16 <= M) {
    // full tile (always taken when M % 16 == 0): unguarded store clause
    float* p = C + (size_t)(mBase + half * 8) * DD + (nBase + l16);
    #pragma unroll
    for (int r = 0; r < 8; ++r) p[(size_t)r * DD] = acc[r] + bv;
  } else {
    #pragma unroll
    for (int r = 0; r < 8; ++r) {
      int gm = mBase + r + half * 8;
      if (gm < M) C[(size_t)gm * DD + nBase + l16] = acc[r] + bv;
    }
  }
}

// ---------------- batchnorm (training-mode, biased var) ----------------

__global__ __launch_bounds__(128) void k_bn_stats(const float* __restrict__ h, int M,
                                                  float* __restrict__ stats) {
  int c = threadIdx.x;              // blockDim = 128, one column per thread
  float s = 0.0f, ss = 0.0f;
  for (int r = blockIdx.x; r < M; r += gridDim.x) {
    float v = h[(size_t)r * DD + c];
    s += v; ss += v * v;
  }
  atomicAdd(&stats[c], s);
  atomicAdd(&stats[DD + c], ss);
}

__global__ void k_bn_apply_relu(float* __restrict__ h, int n, const float* __restrict__ stats,
                                const float* __restrict__ s, const float* __restrict__ b,
                                float invM) {
  int i = blockIdx.x * blockDim.x + threadIdx.x;
  if (i >= n) return;
  int c = i & 127;
  float mu  = stats[c] * invM;
  float var = stats[DD + c] * invM - mu * mu;
  float r   = rsqrtf(var + 1e-5f);
  float v   = (h[i] - mu) * r * s[c] + b[c];
  h[i] = fmaxf(v, 0.0f);
}

// ---------------- graph aggregation (dominant memory phase) ----------------

__global__ void k_agg_init(const float* __restrict__ hin, float* __restrict__ hout,
                           const float* __restrict__ selfnorm, const float* __restrict__ bias,
                           int n) {
  int i = blockIdx.x * blockDim.x + threadIdx.x;
  if (i >= n) return;
  int row = i >> 7, c = i & 127;
  hout[i] = bias[c] + selfnorm[row] * hin[i];
}

// one edge per 32-thread slice, float4 per lane, f32 atomic scatter
__global__ void k_agg_edges(const int* __restrict__ src, const int* __restrict__ dst,
                            const float* __restrict__ norm, const float* __restrict__ hin,
                            float* __restrict__ hout, int E) {
  int t = blockIdx.x * blockDim.x + threadIdx.x;
  int e = t >> 5;
  if (e >= E) return;
  int lane = t & 31;
  int s  = src[e], d = dst[e];
  float nw = norm[e];
  float4 v = ((const float4*)(hin + (size_t)s * DD))[lane];
  float* o = hout + (size_t)d * DD + lane * 4;
  atomicAdd(o + 0, nw * v.x);
  atomicAdd(o + 1, nw * v.y);
  atomicAdd(o + 2, nw * v.z);
  atomicAdd(o + 3, nw * v.w);
}

// ---------------- global mean pool ----------------

__global__ void k_pool_scatter(const float* __restrict__ h, const int* __restrict__ batch,
                               float* __restrict__ pooled, float* __restrict__ cnt, int N) {
  int t = blockIdx.x * blockDim.x + threadIdx.x;
  int row = t >> 5;
  if (row >= N) return;
  int lane = t & 31;
  int g = batch[row];
  float4 v = ((const float4*)(h + (size_t)row * DD))[lane];
  float* o = pooled + (size_t)g * DD + lane * 4;
  atomicAdd(o + 0, v.x);
  atomicAdd(o + 1, v.y);
  atomicAdd(o + 2, v.z);
  atomicAdd(o + 3, v.w);
  if (lane == 0) atomicAdd(&cnt[g], 1.0f);
}

__global__ void k_pool_fin(float* __restrict__ pooled, const float* __restrict__ cnt, int n) {
  int i = blockIdx.x * blockDim.x + threadIdx.x;
  if (i < n) pooled[i] /= fmaxf(cnt[i >> 7], 1.0f);
}

// ---------------- output head: out[g] = dot(gfeat[g,:], out_W) + out_b ----------------

__global__ __launch_bounds__(256) void k_out_head(const float* __restrict__ g,
                                                  const float* __restrict__ out_W,
                                                  const float* __restrict__ out_b,
                                                  float* __restrict__ out, int G) {
  int wave = threadIdx.x >> 5, lane = threadIdx.x & 31;
  int gi = blockIdx.x * 8 + wave;
  if (gi >= G) return;
  float acc = 0.0f;
  #pragma unroll
  for (int c = lane; c < DD; c += 32) acc += g[(size_t)gi * DD + c] * out_W[c];
  #pragma unroll
  for (int off = 16; off > 0; off >>= 1) acc += __shfl_down(acc, off, 32);
  if (lane == 0) out[gi] = acc + out_b[0];
}

// ---------------- host orchestration ----------------

extern "C" void kernel_launch(void* const* d_in, const int* in_sizes, int n_in,
                              void* d_out, int out_size, void* d_ws, size_t ws_size,
                              hipStream_t stream) {
  const float* x         = (const float*)d_in[0];
  const int*   eidx      = (const int*)  d_in[1];
  const float* ew        = (const float*)d_in[2];
  const int*   batch     = (const int*)  d_in[3];
  const float* pre_W     = (const float*)d_in[4];
  const float* pre_b     = (const float*)d_in[5];
  const float* pre_bn_s  = (const float*)d_in[6];
  const float* pre_bn_b  = (const float*)d_in[7];
  const float* convW     = (const float*)d_in[8];
  const float* convb     = (const float*)d_in[9];
  const float* bn_s      = (const float*)d_in[10];
  const float* bn_b      = (const float*)d_in[11];
  const float* post_W    = (const float*)d_in[12];
  const float* post_b    = (const float*)d_in[13];
  const float* post_bn_s = (const float*)d_in[14];
  const float* post_bn_b = (const float*)d_in[15];
  const float* out_W     = (const float*)d_in[16];
  const float* out_b     = (const float*)d_in[17];

  const int N = in_sizes[0] / DD;
  const int E = in_sizes[2];
  const int L = in_sizes[8] / (DD * DD);
  const int G = out_size;

  const int* src = eidx;       // edge_index[0]
  const int* dst = eidx + E;   // edge_index[1]

  // workspace carve (256B aligned)
  char* wp = (char*)d_ws;
  auto carve = [&](size_t bytes) {
    char* r = wp;
    wp += (bytes + 255) & ~(size_t)255;
    return (void*)r;
  };
  float*  dinv     = (float*) carve((size_t)N * 4);
  float*  selfnorm = (float*) carve((size_t)N * 4);
  float*  normw    = (float*) carve((size_t)E * 4);
  float*  bufA     = (float*) carve((size_t)N * DD * 4);
  float*  bufB     = (float*) carve((size_t)N * DD * 4);
  float*  stats    = (float*) carve(2 * DD * 4);
  float*  pooled   = (float*) carve((size_t)G * DD * 4);
  float*  cnt      = (float*) carve((size_t)G * 4);
  float*  gbuf     = (float*) carve((size_t)G * DD * 4);
  __bf16* wpre     = (__bf16*)carve((size_t)DD * DD * 2);
  __bf16* wconv    = (__bf16*)carve((size_t)L * DD * DD * 2);
  __bf16* wpost    = (__bf16*)carve((size_t)DD * DD * 2);
  (void)ws_size; (void)n_in;

  // --- degree / norm ---
  k_fill       <<<CEILDIV(N, 256), 256, 0, stream>>>(dinv, 2.0f, N);      // self-loop weight
  k_scatter_deg<<<CEILDIV(E, 256), 256, 0, stream>>>(dst, ew, dinv, E);
  k_rsqrt      <<<CEILDIV(N, 256), 256, 0, stream>>>(dinv, N);
  k_norm       <<<CEILDIV(E + N, 256), 256, 0, stream>>>(src, dst, ew, dinv, normw, selfnorm, E, N);

  // --- weights -> bf16, pre-swizzled into WMMA B-fragment order ---
  k_cvt_swz<<<CEILDIV(DD * DD, 256), 256, 0, stream>>>(pre_W, wpre);
  for (int l = 0; l < L; ++l)
    k_cvt_swz<<<CEILDIV(DD * DD, 256), 256, 0, stream>>>(convW + (size_t)l * DD * DD,
                                                         wconv + (size_t)l * DD * DD);
  k_cvt_swz<<<CEILDIV(DD * DD, 256), 256, 0, stream>>>(post_W, wpost);

  // --- pre FC -> BN -> ReLU ---
  k_gemm_wmma<<<CEILDIV(N, 16), 256, 0, stream>>>(x, wpre, bufA, pre_b, N);
  k_fill<<<1, 256, 0, stream>>>(stats, 0.0f, 2 * DD);
  k_bn_stats<<<256, 128, 0, stream>>>(bufA, N, stats);
  k_bn_apply_relu<<<CEILDIV(N * DD, 256), 256, 0, stream>>>(bufA, N * DD, stats,
                                                            pre_bn_s, pre_bn_b, 1.0f / (float)N);

  // --- GCNConv stack ---
  for (int l = 0; l < L; ++l) {
    k_gemm_wmma<<<CEILDIV(N, 16), 256, 0, stream>>>(bufA, wconv + (size_t)l * DD * DD,
                                                    bufB, nullptr, N);
    k_agg_init<<<CEILDIV(N * DD, 256), 256, 0, stream>>>(bufB, bufA, selfnorm,
                                                         convb + (size_t)l * DD, N * DD);
    k_agg_edges<<<CEILDIV(E * 32, 256), 256, 0, stream>>>(src, dst, normw, bufB, bufA, E);
    k_fill<<<1, 256, 0, stream>>>(stats, 0.0f, 2 * DD);
    k_bn_stats<<<256, 128, 0, stream>>>(bufA, N, stats);
    k_bn_apply_relu<<<CEILDIV(N * DD, 256), 256, 0, stream>>>(bufA, N * DD, stats,
                                                              bn_s + (size_t)l * DD,
                                                              bn_b + (size_t)l * DD,
                                                              1.0f / (float)N);
  }

  // --- global mean pool ---
  k_fill<<<CEILDIV(G * DD, 256), 256, 0, stream>>>(pooled, 0.0f, G * DD);
  k_fill<<<1, 256, 0, stream>>>(cnt, 0.0f, G);
  k_pool_scatter<<<CEILDIV(N * 32, 256), 256, 0, stream>>>(bufA, batch, pooled, cnt, N);
  k_pool_fin<<<CEILDIV(G * DD, 256), 256, 0, stream>>>(pooled, cnt, G * DD);

  // --- post FC -> BN -> ReLU -> out head ---
  k_gemm_wmma<<<CEILDIV(G, 16), 256, 0, stream>>>(pooled, wpost, gbuf, post_b, G);
  k_fill<<<1, 256, 0, stream>>>(stats, 0.0f, 2 * DD);
  k_bn_stats<<<256, 128, 0, stream>>>(gbuf, G, stats);
  k_bn_apply_relu<<<CEILDIV(G * DD, 256), 256, 0, stream>>>(gbuf, G * DD, stats,
                                                            post_bn_s, post_bn_b, 1.0f / (float)G);
  k_out_head<<<CEILDIV(G, 8), 256, 0, stream>>>(gbuf, out_W, out_b, (float*)d_out, G);
}
